// TemporalDebertaAggregator_20237885898764
// MI455X (gfx1250) — matrix operand order repaired and checked by reading, not compiled, not run on hardware
//
#include <hip/hip_runtime.h>
#include <math.h>

// ---------------------------------------------------------------------------
// DeBERTa-v2 encoder forward for MI455X (gfx1250, wave32, WMMA bf16 + TDM).
//   B=16 S=512 D=768 H=12 Dh=64 F=3072 L=4 span=512
// GEMMs: v_wmma_f32_16x16x32_bf16, weights pre-transposed to [N][K] bf16 so
// all LDS staging is contiguous 2D tiles -> serviced by the Tensor Data Mover
// (tensor_load_to_lds + s_wait_tensorcnt) with vector-load fallback.
// Attention fused per (b,h,16-query tile): QK^T (WMMA) -> Toeplitz c2p/p2c
// bias -> masked softmax in LDS -> P@V (WMMA, V pre-transposed per head).
// ---------------------------------------------------------------------------

#define B_  16
#define S_  512
#define D_  768
#define H_  12
#define DH_ 64
#define F_  3072
#define L_  4
#define QT  16          // queries per attention block
#define NKT 32          // S_/16 key tiles

typedef __bf16 bf16_t;
typedef bf16_t v16bf __attribute__((ext_vector_type(16)));
typedef bf16_t v8bf  __attribute__((ext_vector_type(8)));
typedef float  v8f   __attribute__((ext_vector_type(8)));
typedef unsigned int u32x4 __attribute__((ext_vector_type(4)));
typedef int          i32x8 __attribute__((ext_vector_type(8)));
typedef int          i32x4 __attribute__((ext_vector_type(4)));

#if defined(__HIP_DEVICE_COMPILE__) && __has_builtin(__builtin_amdgcn_tensor_load_to_lds) && __has_builtin(__builtin_amdgcn_s_wait_tensorcnt)
#define TDM_AVAILABLE 1
#else
#define TDM_AVAILABLE 0
#endif

__device__ __forceinline__ v8f wmma_bf16(v16bf a, v16bf b, v8f c) {
#if defined(__HIP_DEVICE_COMPILE__)
  // (neg_a, A, neg_b, B, c_mod, C, reuse_a, reuse_b)
  return __builtin_amdgcn_wmma_f32_16x16x32_bf16(false, a, false, b, (short)0,
                                                 c, false, false);
#else
  (void)a; (void)b;
  return c;
#endif
}

// A-fragment: two contiguous 8x bf16 (16B) pieces per lane.
__device__ __forceinline__ v16bf frag_cat8(const bf16_t* p0, const bf16_t* p1) {
  v8bf lo = *(const v8bf*)p0;
  v8bf hi = *(const v8bf*)p1;
  v16bf r;
#pragma unroll
  for (int i = 0; i < 8; ++i) { r[i] = lo[i]; r[i + 8] = hi[i]; }
  return r;
}
// B-fragment: one contiguous 16x bf16 (32B) piece per lane.
__device__ __forceinline__ v16bf frag16(const bf16_t* p) {
  return *(const v16bf*)p;
}

// ---------------------------------------------------------------------------
// TDM: DMA one dense 2D bf16 tile (tile_rows x tile_cols, row stride in
// elements) from global memory into LDS (packed row-major). D# built per
// CDNA5 ISA 8.3/8.4: group0 = {count, lds_addr, global_addr, type=2},
// group1 = {data_size=2B, tensor_dim0/1, tile_dim0/1, tensor_dim0_stride}.
// Groups 2/3 zero (2D tensor). 6-arg builtin form (clang-23 / therock HIP).
// Issued once per executing wave (EXEC ignored); complete with tensorcnt.
// ---------------------------------------------------------------------------
__device__ __forceinline__ void tdm_load_tile_2d(const void* gptr,
                                                 unsigned lds_byte_off,
                                                 unsigned tile_rows,
                                                 unsigned tile_cols,
                                                 unsigned row_stride) {
#if TDM_AVAILABLE
  const unsigned long long ga = (unsigned long long)(uintptr_t)gptr;
  u32x4 g0 = {1u,                                     // count=1, no gather
              lds_byte_off,                           // lds_addr
              (unsigned)ga,                           // global_addr[31:0]
              (unsigned)((ga >> 32) & 0x01FFFFFFu) |  // global_addr[56:32]
                  0x80000000u};                       // type=2 ("image")
  i32x8 g1;
  g1[0] = 0x00010000;                                        // data_size=2B
  g1[1] = (int)((row_stride & 0xFFFFu) << 16);               // tensor_dim0 lo
  g1[2] = (int)((row_stride >> 16) & 0xFFFFu) |              // tensor_dim0 hi
          (int)((tile_rows & 0xFFFFu) << 16);                // tensor_dim1 lo
  g1[3] = (int)((tile_rows >> 16) & 0xFFFFu) |               // tensor_dim1 hi
          (int)((tile_cols & 0xFFFFu) << 16);                // tile_dim0
  g1[4] = (int)(tile_rows & 0xFFFFu);                        // tile_dim1
  g1[5] = (int)row_stride;                                   // dim0_stride lo
  g1[6] = 0;                                                 // dim0_stride hi
  g1[7] = 0;
  const i32x4 z4 = {0, 0, 0, 0};
  const i32x8 z8 = {0, 0, 0, 0, 0, 0, 0, 0};
  __builtin_amdgcn_tensor_load_to_lds(g0, g1, z4, z4, z8, 0);
#else
  (void)gptr; (void)lds_byte_off; (void)tile_rows; (void)tile_cols;
  (void)row_stride;
#endif
}

__device__ __forceinline__ void tdm_wait() {
#if TDM_AVAILABLE
  __builtin_amdgcn_s_wait_tensorcnt((short)0);
#endif
}

// ---------------------------------------------------------------------------
// GEMM: out[M,N] = epilogue( A[M,K](bf16) @ Wt[N,K](bf16, pre-transposed)
//                            + bias[N] )
// MODE 0: bf16   MODE 1: GELU->bf16   MODE 2: f32   MODE 3: bf16 V-layout
//   (MODE 3 writes Ob[(b*H + h)*DH + n][s], b=m/S, s=m%S, h=col/64, n=col%64)
// Block: 256 threads (8 waves), tile M=128 x N=64, K step 32.
// ---------------------------------------------------------------------------
template <int MODE>
__global__ __launch_bounds__(256) void gemm_bf16_kernel(
    const bf16_t* __restrict__ A, const bf16_t* __restrict__ Wt,
    const float* __restrict__ bias, bf16_t* __restrict__ Ob,
    float* __restrict__ Of, int M, int N, int K) {
  __shared__ __align__(32) bf16_t sA[128][32];   // [m][k]
  __shared__ __align__(32) bf16_t sB[64][32];    // [n][k]
  const int tid  = threadIdx.x;
  const int lane = tid & 31;
  const int wave = tid >> 5;
  const int l16  = lane & 15;
  const int hs   = (lane < 16) ? 0 : 1;
  const int row0 = blockIdx.x * 128;
  const int col0 = blockIdx.y * 64;

  v8f acc[4];
#pragma unroll
  for (int c = 0; c < 4; ++c) acc[c] = {0.f, 0.f, 0.f, 0.f, 0.f, 0.f, 0.f, 0.f};

  for (int k0 = 0; k0 < K; k0 += 32) {
#if TDM_AVAILABLE
    __syncthreads();   // previous iteration's fragment reads complete
    if (wave == 0) {
      tdm_load_tile_2d(&A[(size_t)row0 * K + k0],
                       (unsigned)(uintptr_t)&sA[0][0], 128, 32, (unsigned)K);
      tdm_load_tile_2d(&Wt[(size_t)col0 * K + k0],
                       (unsigned)(uintptr_t)&sB[0][0], 64, 32, (unsigned)K);
      tdm_wait();
    }
    __syncthreads();
#else
    for (int i = tid; i < 512; i += 256) {       // A: 512 v8bf chunks
      const int r = i >> 2, kc = (i & 3) * 8;
      *(v8bf*)&sA[r][kc] = *(const v8bf*)&A[(size_t)(row0 + r) * K + k0 + kc];
    }
    {                                            // Wt: 256 v8bf chunks
      const int i = tid;
      const int nn = i >> 2, kc = (i & 3) * 8;
      *(v8bf*)&sB[nn][kc] =
          *(const v8bf*)&Wt[(size_t)(col0 + nn) * K + k0 + kc];
    }
    __syncthreads();
#endif

    const bf16_t* ap = &sA[wave * 16 + l16][hs * 16];
    const v16bf afrag = frag_cat8(ap, ap + 8);
#pragma unroll
    for (int c = 0; c < 4; ++c) {
      const v16bf bfrag = frag16(&sB[c * 16 + l16][hs * 16]);
      acc[c] = wmma_bf16(afrag, bfrag, acc[c]);
    }
#if !TDM_AVAILABLE
    __syncthreads();
#endif
  }

#pragma unroll
  for (int c = 0; c < 4; ++c) {
    const int col = col0 + c * 16 + l16;
    const float bv = bias ? bias[col] : 0.f;
#pragma unroll
    for (int i = 0; i < 8; ++i) {
      const int m = row0 + wave * 16 + hs * 8 + i;
      float v = acc[c][i] + bv;
      if (MODE == 1) v = 0.5f * v * (1.f + erff(v * 0.70710678118f));
      if (MODE == 2) {
        Of[(size_t)m * N + col] = v;
      } else if (MODE == 3) {
        const int bb = m >> 9, ss = m & (S_ - 1);
        const int hh = col >> 6, nn = col & 63;
        Ob[(((size_t)bb * H_ + hh) * DH_ + nn) * S_ + ss] = (bf16_t)v;
      } else {
        Ob[(size_t)m * N + col] = (bf16_t)v;
      }
    }
  }
}

// ---------------------------------------------------------------------------
// Fused disentangled attention. One block = (b, h, 16-query tile).
// vT layout: [b][h][n(0..63)][s(0..511)]  (bf16)
// ---------------------------------------------------------------------------
__global__ __launch_bounds__(256) void attention_kernel(
    const bf16_t* __restrict__ qb, const bf16_t* __restrict__ kb,
    const bf16_t* __restrict__ vT, const bf16_t* __restrict__ pkb,
    const bf16_t* __restrict__ pqb, const float* __restrict__ maskw,
    bf16_t* __restrict__ ctxb) {
  __shared__ __align__(32) float  sS[QT][S_];        // 32 KB scores / probs
  __shared__ __align__(32) bf16_t sQ[QT][DH_];       // 2 KB
  __shared__ __align__(32) bf16_t sKV[8][16][DH_];   // 16 KB K staging / V reuse

  const int tid  = threadIdx.x;
  const int lane = tid & 31;
  const int wave = tid >> 5;
  const int l16  = lane & 15;
  const int hs   = (lane < 16) ? 0 : 1;

  const int bid = blockIdx.x;
  const int qt  = bid & (NKT - 1);
  const int h   = (bid >> 5) % H_;
  const int b   = bid / (NKT * H_);
  const int q0  = qt * QT;

  // ---- load Q tile (16x64) ----
#if TDM_AVAILABLE
  if (wave == 0) {
    tdm_load_tile_2d(&qb[((size_t)(b * S_) + q0) * D_ + h * DH_],
                     (unsigned)(uintptr_t)&sQ[0][0], 16, 64, D_);
    tdm_wait();
  }
#else
  for (int i = tid; i < 128; i += 256) {      // 128 v8bf chunks
    const int rr = i >> 3, cc = (i & 7) * 8;
    *(v8bf*)&sQ[rr][cc] =
        *(const v8bf*)&qb[((size_t)(b * S_) + q0 + rr) * D_ + h * DH_ + cc];
  }
#endif
  __syncthreads();

  const v16bf aq0 = frag_cat8(&sQ[l16][hs * 16], &sQ[l16][hs * 16 + 8]);
  const v16bf aq1 = frag_cat8(&sQ[l16][32 + hs * 16], &sQ[l16][32 + hs * 16 + 8]);

  // ---- scores = Q @ K^T (each wave owns one 16-key tile per iteration) ----
  for (int it = 0; it < 4; ++it) {
    const int kt = it * 8 + wave;
#if TDM_AVAILABLE
    // wave-private staging: issue + wait within this wave only
    tdm_load_tile_2d(&kb[((size_t)(b * S_) + kt * 16) * D_ + h * DH_],
                     (unsigned)(uintptr_t)&sKV[wave][0][0], 16, 64, D_);
    tdm_wait();
#else
    for (int i = lane; i < 128; i += 32) {
      const int rr = i >> 3, cc = (i & 7) * 8;
      *(v8bf*)&sKV[wave][rr][cc] =
          *(const v8bf*)&kb[((size_t)(b * S_) + kt * 16 + rr) * D_ + h * DH_ + cc];
    }
    __syncthreads();
#endif
    const v16bf bk0 = frag16(&sKV[wave][l16][hs * 16]);
    const v16bf bk1 = frag16(&sKV[wave][l16][32 + hs * 16]);
    v8f acc = {0.f, 0.f, 0.f, 0.f, 0.f, 0.f, 0.f, 0.f};
    acc = wmma_bf16(aq0, bk0, acc);
    acc = wmma_bf16(aq1, bk1, acc);
#pragma unroll
    for (int i = 0; i < 8; ++i)
      sS[hs * 8 + i][kt * 16 + l16] = acc[i];
#if !TDM_AVAILABLE
    __syncthreads();
#endif
  }
  __syncthreads();

  // ---- Toeplitz relative-position bias + scale + mask ----
  // scores[q,k] += q . pk[q-k+512]  +  k . pq[k-q+512]; then /sqrt(3*Dh), mask
  {
    const float inv_scale = 0.0721687836f;   // 1/sqrt(192)
    const int qq    = tid >> 4;              // 0..15
    const int kbase = (tid & 15) * 32;       // 32 keys per thread
    const int qg    = q0 + qq;
    const float mq  = maskw[b * S_ + qg];
    for (int j = 0; j < 32; ++j) {
      const int k = kbase + j;
      int r1 = qg - k + 512; r1 = r1 < 0 ? 0 : (r1 > 1023 ? 1023 : r1);
      int r2 = k - qg + 512; r2 = r2 < 0 ? 0 : (r2 > 1023 ? 1023 : r2);
      const bf16_t* pkr = pkb + (size_t)r1 * D_ + h * DH_;
      const bf16_t* pqr = pqb + (size_t)r2 * D_ + h * DH_;
      const bf16_t* kvr = kb + ((size_t)(b * S_) + k) * D_ + h * DH_;
      float c2p = 0.f, p2c = 0.f;
#pragma unroll
      for (int d = 0; d < DH_; ++d) {
        c2p += (float)sQ[qq][d] * (float)pkr[d];
        p2c += (float)kvr[d] * (float)pqr[d];
      }
      const float sc = (sS[qq][k] + c2p + p2c) * inv_scale;
      const float mk = maskw[b * S_ + k];
      sS[qq][k] = (mq > 0.f && mk > 0.f) ? sc : -3.0e38f;
    }
  }
  __syncthreads();

  // ---- masked softmax; each wave owns 2 rows ----
  for (int rr = 0; rr < 2; ++rr) {
    const int r = wave * 2 + rr;
    float mx = -3.4e38f;
    for (int c = lane; c < S_; c += 32) mx = fmaxf(mx, sS[r][c]);
#pragma unroll
    for (int off = 16; off > 0; off >>= 1) mx = fmaxf(mx, __shfl_xor(mx, off, 32));
    float sum = 0.f;
    for (int c = lane; c < S_; c += 32) {
      const float e = __expf(sS[r][c] - mx);
      sS[r][c] = e;
      sum += e;
    }
#pragma unroll
    for (int off = 16; off > 0; off >>= 1) sum += __shfl_xor(sum, off, 32);
    const float inv = 1.f / sum;
    const float mq  = maskw[b * S_ + q0 + r];
    for (int c = lane; c < S_; c += 32) {
      const float mk = maskw[b * S_ + c];
      sS[r][c] = (mq > 0.f && mk > 0.f) ? sS[r][c] * inv : 0.f;
    }
  }
  __syncthreads();

  // ---- repack probs f32 -> bf16 in-place (reg staging avoids aliasing) ----
  bf16_t* sP = (bf16_t*)&sS[0][0];
  {
    const int qq = tid >> 4;
    const int kbase = (tid & 15) * 32;
    float tmp[32];
#pragma unroll
    for (int j = 0; j < 32; ++j) tmp[j] = sS[qq][kbase + j];
    __syncthreads();
#pragma unroll
    for (int j = 0; j < 32; ++j) sP[qq * S_ + kbase + j] = (bf16_t)tmp[j];
  }
  __syncthreads();

  // ---- ctx = P(16x512) @ V(512x64); waves 0..3 own 16-col slices ----
  bf16_t (*sVt)[32] = (bf16_t(*)[32]) & sKV[0][0][0];  // [n][k], 64x32
  const bf16_t* vth = vT + ((size_t)b * H_ + h) * DH_ * S_;
  v8f cacc = {0.f, 0.f, 0.f, 0.f, 0.f, 0.f, 0.f, 0.f};
  for (int kc = 0; kc < 16; ++kc) {
#if TDM_AVAILABLE
    if (wave == 0) {
      tdm_load_tile_2d(&vth[kc * 32], (unsigned)(uintptr_t)&sVt[0][0],
                       64, 32, S_);
      tdm_wait();
    }
#else
    {
      const int i = tid;                      // 256 v8bf chunks, one each
      const int nn = i >> 2, c8 = (i & 3) * 8;
      *(v8bf*)&sVt[nn][c8] =
          *(const v8bf*)&vth[(size_t)nn * S_ + kc * 32 + c8];
    }
#endif
    __syncthreads();
    if (wave < 4) {
      const bf16_t* ap = &sP[l16 * S_ + kc * 32 + hs * 16];
      const v16bf af = frag_cat8(ap, ap + 8);
      const v16bf bfr = frag16(&sVt[wave * 16 + l16][hs * 16]);
      cacc = wmma_bf16(af, bfr, cacc);
    }
    __syncthreads();
  }
  if (wave < 4) {
#pragma unroll
    for (int i = 0; i < 8; ++i) {
      const int m = hs * 8 + i;
      ctxb[((size_t)(b * S_) + q0 + m) * D_ + h * DH_ + wave * 16 + l16] =
          (bf16_t)cacc[i];
    }
  }
}

// ---------------------------------------------------------------------------
// LayerNorm(x + residual) -> f32 out + bf16 out. One block per row (D=768).
// ---------------------------------------------------------------------------
__global__ __launch_bounds__(256) void ln_residual_kernel(
    const float* __restrict__ X, const float* __restrict__ R,
    const float* __restrict__ gamma, const float* __restrict__ beta,
    float* __restrict__ Y, bf16_t* __restrict__ Yb) {
  __shared__ float rs[256];
  __shared__ float rq[256];
  const int row = blockIdx.x;
  const int tid = threadIdx.x;
  const float* x = X + (size_t)row * D_;
  const float* r = R + (size_t)row * D_;
  float v[3];
  float s = 0.f, sq = 0.f;
#pragma unroll
  for (int j = 0; j < 3; ++j) {
    const int idx = tid + j * 256;
    v[j] = x[idx] + r[idx];
    s += v[j];
    sq += v[j] * v[j];
  }
  rs[tid] = s; rq[tid] = sq;
  __syncthreads();
  for (int off = 128; off > 0; off >>= 1) {
    if (tid < off) { rs[tid] += rs[tid + off]; rq[tid] += rq[tid + off]; }
    __syncthreads();
  }
  const float mean = rs[0] * (1.f / D_);
  const float var  = rq[0] * (1.f / D_) - mean * mean;
  const float inv  = rsqrtf(var + 1e-7f);
#pragma unroll
  for (int j = 0; j < 3; ++j) {
    const int idx = tid + j * 256;
    const float y = (v[j] - mean) * inv * gamma[idx] + beta[idx];
    Y[(size_t)row * D_ + idx]  = y;
    Yb[(size_t)row * D_ + idx] = (bf16_t)y;
  }
}

// ---------------------------------------------------------------------------
// Helpers: converts, transposed weight downcast, CLS insertion, mask, gather
// ---------------------------------------------------------------------------
__global__ __launch_bounds__(256) void cvt_f32_bf16_kernel(
    const float* __restrict__ in, bf16_t* __restrict__ out, long n) {
  const long i = (long)blockIdx.x * 256 + threadIdx.x;
  if (i < n) out[i] = (bf16_t)in[i];
}

// in: (L, K, N) f32  ->  out: (L, N, K) bf16
__global__ __launch_bounds__(256) void cvt_transpose_kernel(
    const float* __restrict__ in, bf16_t* __restrict__ out, int K, int N,
    long total) {
  const long i = (long)blockIdx.x * 256 + threadIdx.x;
  if (i >= total) return;
  const long kn = (long)K * N;
  const int l = (int)(i / kn);
  const long rem = i - (long)l * kn;
  const int k = (int)(rem / N);
  const int n = (int)(rem - (long)k * N);
  out[(size_t)l * kn + (size_t)n * K + k] = (bf16_t)in[i];
}

__global__ __launch_bounds__(256) void prep_h_kernel(
    const float* __restrict__ x, const float* __restrict__ cls,
    const int* __restrict__ last, float* __restrict__ h) {
  const long i = (long)blockIdx.x * 256 + threadIdx.x;
  if (i >= (long)B_ * S_ * D_) return;
  const int d = (int)(i % D_);
  const int s = (int)((i / D_) % S_);
  const int b = (int)(i / ((long)S_ * D_));
  int fp = last[b] + 1;
  if (fp > S_ - 1) fp = S_ - 1;
  h[i] = (s == fp) ? cls[d] : x[i];
}

__global__ __launch_bounds__(256) void prep_mask_kernel(
    const float* __restrict__ mask, const int* __restrict__ last,
    float* __restrict__ mw) {
  const int i = blockIdx.x * 256 + threadIdx.x;
  if (i >= B_ * S_) return;
  const int b = i / S_;
  const int s = i % S_;
  int fp = last[b] + 1;
  if (fp > S_ - 1) fp = S_ - 1;
  mw[i] = (s == fp) ? 1.f : mask[i];
}

__global__ __launch_bounds__(256) void gather_out_kernel(
    const float* __restrict__ h, const int* __restrict__ last,
    float* __restrict__ out) {
  const int i = blockIdx.x * 256 + threadIdx.x;
  if (i >= B_ * D_) return;
  const int b = i / D_;
  const int d = i % D_;
  int fp = last[b] + 1;
  if (fp > S_ - 1) fp = S_ - 1;
  out[i] = h[((size_t)b * S_ + fp) * D_ + d];
}

// ---------------------------------------------------------------------------
// Host orchestration
// ---------------------------------------------------------------------------
extern "C" void kernel_launch(void* const* d_in, const int* in_sizes, int n_in,
                              void* d_out, int out_size, void* d_ws,
                              size_t ws_size, hipStream_t stream) {
  (void)in_sizes; (void)n_in; (void)out_size; (void)ws_size;

  // setup_inputs() dict order, params flattened in insertion order.
  const float* x    = (const float*)d_in[0];
  const float* mask = (const float*)d_in[1];
  const int*   last = (const int*)d_in[2];
  const float* cls  = (const float*)d_in[3];
  const float* rel  = (const float*)d_in[4];
  const float* Wq = (const float*)d_in[5];   const float* bq  = (const float*)d_in[6];
  const float* Wk = (const float*)d_in[7];   const float* bk  = (const float*)d_in[8];
  const float* Wv = (const float*)d_in[9];   const float* bv  = (const float*)d_in[10];
  const float* Wpk = (const float*)d_in[11]; const float* bpk = (const float*)d_in[12];
  const float* Wpq = (const float*)d_in[13]; const float* bpq = (const float*)d_in[14];
  const float* Wo = (const float*)d_in[15];  const float* bo  = (const float*)d_in[16];
  const float* g1 = (const float*)d_in[17];  const float* b1  = (const float*)d_in[18];
  const float* Wi = (const float*)d_in[19];  const float* bi  = (const float*)d_in[20];
  const float* Wf = (const float*)d_in[21];  const float* bfv = (const float*)d_in[22];
  const float* g2 = (const float*)d_in[23];  const float* b2  = (const float*)d_in[24];

  // --- carve workspace (~270 MB) ---
  char* base = (char*)d_ws;
  size_t off = 0;
  auto carve = [&](size_t bytes) -> void* {
    void* p = base + off;
    off = (off + bytes + 255) & ~(size_t)255;
    return p;
  };
  const size_t nAct = (size_t)B_ * S_ * D_;          // 6.29M
  float*  h      = (float*)carve(nAct * 4);
  float*  attnF  = (float*)carve(nAct * 4);
  float*  t1     = (float*)carve(nAct * 4);
  bf16_t* hb     = (bf16_t*)carve(nAct * 2);
  bf16_t* qbuf   = (bf16_t*)carve(nAct * 2);
  bf16_t* kbuf   = (bf16_t*)carve(nAct * 2);
  bf16_t* vTbuf  = (bf16_t*)carve(nAct * 2);          // [b][h][n][s]
  bf16_t* ctxb   = (bf16_t*)carve(nAct * 2);
  bf16_t* attnb  = (bf16_t*)carve(nAct * 2);
  bf16_t* ffb    = (bf16_t*)carve((size_t)B_ * S_ * F_ * 2);
  float*  maskw  = (float*)carve((size_t)B_ * S_ * 4);
  bf16_t* relb   = (bf16_t*)carve((size_t)1024 * D_ * 2);
  bf16_t* pkb    = (bf16_t*)carve((size_t)1024 * D_ * 2);
  bf16_t* pqb    = (bf16_t*)carve((size_t)1024 * D_ * 2);
  bf16_t* Wqb  = (bf16_t*)carve((size_t)L_ * D_ * D_ * 2);   // all transposed
  bf16_t* Wkb  = (bf16_t*)carve((size_t)L_ * D_ * D_ * 2);
  bf16_t* Wvb  = (bf16_t*)carve((size_t)L_ * D_ * D_ * 2);
  bf16_t* Wpkb = (bf16_t*)carve((size_t)L_ * D_ * D_ * 2);
  bf16_t* Wpqb = (bf16_t*)carve((size_t)L_ * D_ * D_ * 2);
  bf16_t* Wob  = (bf16_t*)carve((size_t)L_ * D_ * D_ * 2);
  bf16_t* Wib  = (bf16_t*)carve((size_t)L_ * D_ * F_ * 2);
  bf16_t* Wfb  = (bf16_t*)carve((size_t)L_ * F_ * D_ * 2);

  auto cvtT = [&](const float* src, bf16_t* dst, int K, int N) {
    const long n = (long)L_ * K * N;
    cvt_transpose_kernel<<<(unsigned)((n + 255) / 256), 256, 0, stream>>>(
        src, dst, K, N, n);
  };

  // one-time weight downcast + transpose to [N][K] (L2-resident thereafter)
  cvtT(Wq, Wqb, D_, D_);
  cvtT(Wk, Wkb, D_, D_);
  cvtT(Wv, Wvb, D_, D_);
  cvtT(Wpk, Wpkb, D_, D_);
  cvtT(Wpq, Wpqb, D_, D_);
  cvtT(Wo, Wob, D_, D_);
  cvtT(Wi, Wib, D_, F_);
  cvtT(Wf, Wfb, F_, D_);
  cvt_f32_bf16_kernel<<<(unsigned)(((long)1024 * D_ + 255) / 256), 256, 0,
                        stream>>>(rel, relb, (long)1024 * D_);

  prep_h_kernel<<<(unsigned)((nAct + 255) / 256), 256, 0, stream>>>(x, cls,
                                                                    last, h);
  prep_mask_kernel<<<(B_ * S_ + 255) / 256, 256, 0, stream>>>(mask, last,
                                                              maskw);
  cvt_f32_bf16_kernel<<<(unsigned)((nAct + 255) / 256), 256, 0, stream>>>(
      h, hb, (long)nAct);

  const int M = B_ * S_;                 // 8192
  const dim3 gD(M / 128, D_ / 64);       // 64 x 12
  const dim3 gP(1024 / 128, D_ / 64);    // 8 x 12
  const dim3 gF(M / 128, F_ / 64);       // 64 x 48

  for (int l = 0; l < L_; ++l) {
    const size_t wOff = (size_t)l * D_ * D_;
    const size_t wOffI = (size_t)l * D_ * F_;
    const size_t bOff = (size_t)l * D_;
    const size_t bOffF = (size_t)l * F_;

    gemm_bf16_kernel<0><<<gD, 256, 0, stream>>>(hb, Wqb + wOff, bq + bOff,
                                                qbuf, nullptr, M, D_, D_);
    gemm_bf16_kernel<0><<<gD, 256, 0, stream>>>(hb, Wkb + wOff, bk + bOff,
                                                kbuf, nullptr, M, D_, D_);
    gemm_bf16_kernel<3><<<gD, 256, 0, stream>>>(hb, Wvb + wOff, bv + bOff,
                                                vTbuf, nullptr, M, D_, D_);
    gemm_bf16_kernel<0><<<gP, 256, 0, stream>>>(relb, Wpkb + wOff, bpk + bOff,
                                                pkb, nullptr, 1024, D_, D_);
    gemm_bf16_kernel<0><<<gP, 256, 0, stream>>>(relb, Wpqb + wOff, bpq + bOff,
                                                pqb, nullptr, 1024, D_, D_);

    attention_kernel<<<B_ * H_ * NKT, 256, 0, stream>>>(qbuf, kbuf, vTbuf,
                                                        pkb, pqb, maskw, ctxb);

    gemm_bf16_kernel<2><<<gD, 256, 0, stream>>>(ctxb, Wob + wOff, bo + bOff,
                                                nullptr, t1, M, D_, D_);
    ln_residual_kernel<<<M, 256, 0, stream>>>(t1, h, g1 + bOff, b1 + bOff,
                                              attnF, attnb);

    gemm_bf16_kernel<1><<<gF, 256, 0, stream>>>(attnb, Wib + wOffI, bi + bOffF,
                                                ffb, nullptr, M, F_, D_);
    gemm_bf16_kernel<2><<<gD, 256, 0, stream>>>(ffb, Wfb + wOffI, bfv + bOff,
                                                nullptr, t1, M, D_, F_);
    ln_residual_kernel<<<M, 256, 0, stream>>>(t1, attnF, g2 + bOff, b2 + bOff,
                                              h, hb);
  }

  gather_out_kernel<<<(B_ * D_ + 255) / 256, 256, 0, stream>>>(h, last,
                                                               (float*)d_out);
}
